// GemNetPotential_43361989820925
// MI455X (gfx1250) — compile-verified, matrix operand order checked by reading, not run on hardware
//
#include <hip/hip_runtime.h>
#include <hip/hip_bf16.h>

#define Bb 4
#define Nn 192
#define Hh 128
#define NBb 3
#define NRr 6
#define NSs 4
#define CUTOFF 5.0f
#define WIDTHc (5.0f / 6.0f)

typedef __attribute__((ext_vector_type(16))) _Float16 v16h;
typedef __attribute__((ext_vector_type(8)))  float    v8f;

// fast silu: x * rcp(1 + exp(-x)) -> v_exp_f32 + v_add + v_rcp_f32 + v_mul
__device__ __forceinline__ float silu(float x) {
  return x * __builtin_amdgcn_rcpf(1.0f + __expf(-x));
}

// ---------------- f = emb[node_indices] ----------------
__global__ void embed_kernel(const int* __restrict__ idx,
                             const float* __restrict__ emb,
                             float* __restrict__ f) {
  int row = blockIdx.x;          // b*N+i, 768 rows
  int t = threadIdx.x;           // 128
  f[row * Hh + t] = emb[idx[row] * Hh + t];
}

// ---------------- per-edge geometry: em, rbf(6), dir_feat(4) ----------------
__global__ void geom_kernel(const float* __restrict__ pos,
                            const int* __restrict__ adj,
                            const float* __restrict__ mask,
                            const float* __restrict__ dpw,  // [3,NS]
                            const float* __restrict__ dpb,  // [NS]
                            float* __restrict__ em,
                            float* __restrict__ rbf,
                            float* __restrict__ dirf) {
  int e = blockIdx.x * 256 + threadIdx.x;      // E = 147456 = 576*256
  int b = e / (Nn * Nn);
  int r = e - b * (Nn * Nn);
  int i = r / Nn;
  int j = r - i * Nn;
  const float* pi = pos + (b * Nn + i) * 3;
  const float* pj = pos + (b * Nn + j) * 3;
  float dx = pi[0] - pj[0], dy = pi[1] - pj[1], dz = pi[2] - pj[2];
  float d = sqrtf(dx * dx + dy * dy + dz * dz);
  bool ok = (mask[b * Nn + i] > 0.0f) && (mask[b * Nn + j] > 0.0f) &&
            (d <= CUTOFF) && (adj[e] > 0) && (i != j);
  float e_m = ok ? 1.0f : 0.0f;
  em[e] = e_m;
  float inv = e_m * __builtin_amdgcn_rcpf(d + 1e-8f);
  float ux = dx * inv, uy = dy * inv, uz = dz * inv;
  const float iw = 1.0f / (WIDTHc + 1e-8f);
#pragma unroll
  for (int k = 0; k < NRr; ++k) {
    float z = (d - (float)k) * iw;           // centers = 0..5, spacing 1
    rbf[e * NRr + k] = __expf(-0.5f * z * z) * e_m;
  }
#pragma unroll
  for (int s = 0; s < NSs; ++s) {
    float v = ux * dpw[0 * NSs + s] + uy * dpw[1 * NSs + s] +
              uz * dpw[2 * NSs + s] + dpb[s];
    dirf[e * NSs + s] = tanhf(v);
  }
}

// ---------------- per-block weight fusion + f16 K-pair packing ----------------
// Wp rows 0..127 = rw2 @ ew1[256:384], rows 128..255 = dw2 @ ew1[384:512]
// pack layout: half index ((k>>1)*Hh + n)*2 + (k&1)  -> dword = (K, K+1)
__global__ void prep_kernel(const float* __restrict__ rw2,
                            const float* __restrict__ dw2,
                            const float* __restrict__ ew1,
                            const float* __restrict__ ew2,
                            const float* __restrict__ rb2,
                            const float* __restrict__ db2,
                            const float* __restrict__ eb1,
                            _Float16* __restrict__ Wp,
                            _Float16* __restrict__ e2p,
                            float* __restrict__ biasF) {
  int n = threadIdx.x;    // 128
  int bk = blockIdx.x;    // 385
  if (bk < 256) {
    int k = bk;
    float s = 0.0f;
    if (k < Hh) {
      for (int c = 0; c < Hh; ++c) s += rw2[k * Hh + c] * ew1[(2 * Hh + c) * Hh + n];
    } else {
      int kk = k - Hh;
      for (int c = 0; c < Hh; ++c) s += dw2[kk * Hh + c] * ew1[(3 * Hh + c) * Hh + n];
    }
    Wp[((k >> 1) * Hh + n) * 2 + (k & 1)] = (_Float16)s;
  } else if (bk < 384) {
    int k = bk - 256;
    e2p[((k >> 1) * Hh + n) * 2 + (k & 1)] = (_Float16)ew2[k * Hh + n];
  } else {
    float s = eb1[n];
    for (int c = 0; c < Hh; ++c)
      s += rb2[c] * ew1[(2 * Hh + c) * Hh + n] + db2[c] * ew1[(3 * Hh + c) * Hh + n];
    biasF[n] = s;
  }
}

// ---------------- fc = f@ew1[0:128], fn = f@ew1[128:256] ----------------
__global__ void node_proj_kernel(const float* __restrict__ f,
                                 const float* __restrict__ ew1,
                                 float* __restrict__ fc,
                                 float* __restrict__ fn) {
  __shared__ float sf[Hh];
  int row = blockIdx.x;    // 768
  int t = threadIdx.x;     // 256
  if (t < Hh) sf[t] = f[row * Hh + t];
  __syncthreads();
  float s = 0.0f;
  if (t < Hh) {
    for (int c = 0; c < Hh; ++c) s += sf[c] * ew1[c * Hh + t];
    fc[row * Hh + t] = s;
  } else {
    int n = t - Hh;
    for (int c = 0; c < Hh; ++c) s += sf[c] * ew1[(Hh + c) * Hh + n];
    fn[row * Hh + n] = s;
  }
}

// ---------------- big fused edge-message kernel (WMMA) ----------------
// One workgroup per (b,i) row: 8 waves, wave w owns output cols [16w,16w+16)
__global__ __launch_bounds__(256)
void edge_kernel(const float* __restrict__ emG,
                 const float* __restrict__ rbfG,
                 const float* __restrict__ dirG,
                 const float* __restrict__ fcG,
                 const float* __restrict__ fnG,
                 float* __restrict__ aggG,
                 const unsigned* __restrict__ Wp,    // [256/2][128] dwords
                 const unsigned* __restrict__ e2p,   // [128/2][128] dwords
                 const float* __restrict__ biasF,
                 const float* __restrict__ rw1,      // [6,128]
                 const float* __restrict__ rb1,      // [128]
                 const float* __restrict__ dw1,      // [4,128]
                 const float* __restrict__ db1,      // [128]
                 const float* __restrict__ eb2) {    // [128]
  __shared__ float   s_em[Nn];
  __shared__ float   s_rbf[Nn * NRr];
  __shared__ float   s_dir[Nn * NSs];
  __shared__ float   s_fc[Hh];
  __shared__ float   s_bias[Hh];
  __shared__ float   s_w1[(NRr + NSs) * Hh];
  __shared__ float   s_b1[2 * Hh];
  __shared__ float   s_fn[16 * Hh];
  __shared__ _Float16 s_rd1[16 * 256];
  __shared__ _Float16 s_h[16 * Hh];

  const int tid = threadIdx.x;
  const int lane = tid & 31;
  const int wave = tid >> 5;
  const int bi = blockIdx.x;                 // b*N + i
  const int bB = (bi / Nn) * Nn;             // b*N
  const long e0 = (long)bi * Nn;

  for (int x = tid; x < Nn; x += 256) s_em[x] = emG[e0 + x];
  for (int x = tid; x < Nn * NRr; x += 256) s_rbf[x] = rbfG[e0 * NRr + x];
  for (int x = tid; x < Nn * NSs; x += 256) s_dir[x] = dirG[e0 * NSs + x];
  if (tid < Hh) { s_fc[tid] = fcG[bi * Hh + tid]; s_bias[tid] = biasF[tid]; }
  for (int x = tid; x < NRr * Hh; x += 256) s_w1[x] = rw1[x];
  for (int x = tid; x < NSs * Hh; x += 256) s_w1[NRr * Hh + x] = dw1[x];
  if (tid < Hh) s_b1[tid] = rb1[tid];
  else          s_b1[tid] = db1[tid - Hh];
  __syncthreads();

  const int nloc = lane & 15;
  const int hi = lane >> 4;
  const int ncol = wave * 16 + nloc;
  const float eb2v = eb2[ncol];
  const float fcbias = s_fc[ncol] + s_bias[ncol];   // loop-invariant C init term

  float aggv[8];
#pragma unroll
  for (int k = 0; k < 8; ++k) aggv[k] = 0.0f;

  for (int mt = 0; mt < Nn / 16; ++mt) {
    const int jb = mt * 16;
    __syncthreads();
    // r1|d1 tile (16 rows x 256 K) -> LDS f16, silu applied
    for (int idx = tid; idx < 16 * 256; idx += 256) {
      int r = idx >> 8, k = idx & 255;
      float s;
      if (k < Hh) {
        s = s_b1[k];
#pragma unroll
        for (int c = 0; c < NRr; ++c) s += s_rbf[(jb + r) * NRr + c] * s_w1[c * Hh + k];
      } else {
        int kk = k - Hh;
        s = s_b1[Hh + kk];
#pragma unroll
        for (int c = 0; c < NSs; ++c) s += s_dir[(jb + r) * NSs + c] * s_w1[(NRr + c) * Hh + kk];
      }
      s_rd1[idx] = (_Float16)silu(s);
    }
    // fn tile (16 rows x 128)
    for (int idx = tid; idx < 16 * Hh; idx += 256) {
      int r = idx >> 7, c = idx & 127;
      s_fn[idx] = fnG[(bB + jb + r) * Hh + c];
    }
    __syncthreads();

    // GEMM1: h_pre = fc + fn + bias + rd1 @ Wcat   (K = 256)
    v8f acc;
#pragma unroll
    for (int k = 0; k < 8; ++k)
      acc[k] = fcbias + s_fn[(k + hi * 8) * Hh + ncol];
#pragma unroll
    for (int ks = 0; ks < 8; ++ks) {
      union { v16h v; unsigned u[8]; } af, bf;
#pragma unroll
      for (int r8 = 0; r8 < 8; ++r8) {
        int k = ks * 32 + ((r8 & 4) << 2) + hi * 8 + (r8 & 3) * 2;
        af.u[r8] = *(const unsigned*)&s_rd1[nloc * 256 + k];
        bf.u[r8] = Wp[(k >> 1) * Hh + ncol];
      }
      acc = __builtin_amdgcn_wmma_f32_16x16x32_f16(false, af.v, false, bf.v,
                                                   (short)0, acc, false, false);
    }
    // silu -> stage h tile in LDS (f16)
#pragma unroll
    for (int k = 0; k < 8; ++k)
      s_h[(k + hi * 8) * Hh + ncol] = (_Float16)silu(acc[k]);
    __syncthreads();

    // GEMM2: m = h @ ew2 + eb2   (K = 128)
    v8f acc2;
#pragma unroll
    for (int k = 0; k < 8; ++k) acc2[k] = eb2v;
#pragma unroll
    for (int ks = 0; ks < 4; ++ks) {
      union { v16h v; unsigned u[8]; } af, bf;
#pragma unroll
      for (int r8 = 0; r8 < 8; ++r8) {
        int k = ks * 32 + ((r8 & 4) << 2) + hi * 8 + (r8 & 3) * 2;
        af.u[r8] = *(const unsigned*)&s_h[nloc * Hh + k];
        bf.u[r8] = e2p[(k >> 1) * Hh + ncol];
      }
      acc2 = __builtin_amdgcn_wmma_f32_16x16x32_f16(false, af.v, false, bf.v,
                                                    (short)0, acc2, false, false);
    }
    // mask by em and accumulate the j-reduction inside the C fragment
#pragma unroll
    for (int k = 0; k < 8; ++k)
      aggv[k] += acc2[k] * s_em[jb + k + hi * 8];
  }

  // fragment rows -> per-column sum: sum 8 VGPRs then fold lane halves
  float s = 0.0f;
#pragma unroll
  for (int k = 0; k < 8; ++k) s += aggv[k];
  float tot = s + __shfl(s, nloc + 16, 32);
  if (lane < 16) aggG[bi * Hh + ncol] = tot;
}

// ---------------- node update: f += MLP([f, agg]) ----------------
__global__ void node_update_kernel(float* __restrict__ f,
                                   const float* __restrict__ agg,
                                   const float* __restrict__ nw1,
                                   const float* __restrict__ nb1,
                                   const float* __restrict__ nw2,
                                   const float* __restrict__ nb2) {
  __shared__ float sx[2 * Hh];
  __shared__ float su[Hh];
  int row = blockIdx.x;     // 768
  int t = threadIdx.x;      // 128
  sx[t] = f[row * Hh + t];
  sx[Hh + t] = agg[row * Hh + t];
  __syncthreads();
  float s = nb1[t];
  for (int c = 0; c < 2 * Hh; ++c) s += sx[c] * nw1[c * Hh + t];
  su[t] = silu(s);
  __syncthreads();
  float u = nb2[t];
  for (int c = 0; c < Hh; ++c) u += su[c] * nw2[c * Hh + t];
  f[row * Hh + t] = sx[t] + u;
}

// ---------------- output head + masked batch reduction ----------------
__global__ void output_kernel(const float* __restrict__ f,
                              const float* __restrict__ mask,
                              const float* __restrict__ w1,
                              const float* __restrict__ b1,
                              const float* __restrict__ w2,
                              const float* __restrict__ b2,
                              float* __restrict__ out) {
  __shared__ float sf[Hh];
  __shared__ float sred[Hh];
  int b = blockIdx.x;       // 4
  int t = threadIdx.x;      // 128
  float part = 0.0f, msum = 0.0f;
  for (int i = 0; i < Nn; ++i) {
    __syncthreads();
    sf[t] = f[(b * Nn + i) * Hh + t];
    __syncthreads();
    float s = b1[t];
    for (int c = 0; c < Hh; ++c) s += sf[c] * w1[c * Hh + t];
    float mi = mask[b * Nn + i];
    part += silu(s) * w2[t] * mi;
    msum += mi;
  }
  sred[t] = part;
  __syncthreads();
  for (int off = 64; off > 0; off >>= 1) {
    if (t < off) sred[t] += sred[t + off];
    __syncthreads();
  }
  if (t == 0) out[b] = sred[0] + b2[0] * msum;
}

extern "C" void kernel_launch(void* const* d_in, const int* in_sizes, int n_in,
                              void* d_out, int out_size, void* d_ws, size_t ws_size,
                              hipStream_t stream) {
  const int*   node_idx = (const int*)d_in[0];
  const float* pos      = (const float*)d_in[1];
  const int*   adj      = (const int*)d_in[2];
  const float* mask     = (const float*)d_in[3];
  const float* emb      = (const float*)d_in[4];
  const float* dpw      = (const float*)d_in[5];
  const float* dpb      = (const float*)d_in[6];
  const float* rbf_w1   = (const float*)d_in[7];
  const float* rbf_b1   = (const float*)d_in[8];
  const float* rbf_w2   = (const float*)d_in[9];
  const float* rbf_b2   = (const float*)d_in[10];
  const float* dir_w1   = (const float*)d_in[11];
  const float* dir_b1   = (const float*)d_in[12];
  const float* dir_w2   = (const float*)d_in[13];
  const float* dir_b2   = (const float*)d_in[14];
  const float* edge_w1  = (const float*)d_in[15];
  const float* edge_b1  = (const float*)d_in[16];
  const float* edge_w2  = (const float*)d_in[17];
  const float* edge_b2  = (const float*)d_in[18];
  const float* node_w1  = (const float*)d_in[19];
  const float* node_b1  = (const float*)d_in[20];
  const float* node_w2  = (const float*)d_in[21];
  const float* node_b2  = (const float*)d_in[22];
  const float* out_w1   = (const float*)d_in[23];
  const float* out_b1   = (const float*)d_in[24];
  const float* out_w2   = (const float*)d_in[25];
  const float* out_b2   = (const float*)d_in[26];
  float* out = (float*)d_out;

  const int ROWS = Bb * Nn;              // 768
  const int E = Bb * Nn * Nn;            // 147456

  // workspace layout
  float* w_f    = (float*)d_ws;                          // [768,128]
  float* w_agg  = w_f    + ROWS * Hh;
  float* w_fc   = w_agg  + ROWS * Hh;
  float* w_fn   = w_fc   + ROWS * Hh;
  float* w_em   = w_fn   + ROWS * Hh;                    // [E]
  float* w_rbf  = w_em   + E;                            // [E,6]
  float* w_dir  = w_rbf  + E * NRr;                      // [E,4]
  float* w_bias = w_dir  + E * NSs;                      // [128]
  _Float16* w_Wp  = (_Float16*)(w_bias + Hh);            // [128][128] dword pairs
  _Float16* w_e2p = w_Wp + 256 * Hh;                     // [64][128] dword pairs

  embed_kernel<<<dim3(ROWS), dim3(Hh), 0, stream>>>(node_idx, emb, w_f);
  geom_kernel<<<dim3(E / 256), dim3(256), 0, stream>>>(pos, adj, mask, dpw, dpb,
                                                       w_em, w_rbf, w_dir);

  for (int t = 0; t < NBb; ++t) {
    const float* ew1_t = edge_w1 + (size_t)t * 4 * Hh * Hh;
    prep_kernel<<<dim3(385), dim3(Hh), 0, stream>>>(
        rbf_w2 + (size_t)t * Hh * Hh, dir_w2 + (size_t)t * Hh * Hh, ew1_t,
        edge_w2 + (size_t)t * Hh * Hh, rbf_b2 + t * Hh, dir_b2 + t * Hh,
        edge_b1 + t * Hh, w_Wp, w_e2p, w_bias);
    node_proj_kernel<<<dim3(ROWS), dim3(256), 0, stream>>>(w_f, ew1_t, w_fc, w_fn);
    edge_kernel<<<dim3(ROWS), dim3(256), 0, stream>>>(
        w_em, w_rbf, w_dir, w_fc, w_fn, w_agg,
        (const unsigned*)w_Wp, (const unsigned*)w_e2p, w_bias,
        rbf_w1 + (size_t)t * NRr * Hh, rbf_b1 + t * Hh,
        dir_w1 + (size_t)t * NSs * Hh, dir_b1 + t * Hh,
        edge_b2 + t * Hh);
    node_update_kernel<<<dim3(ROWS), dim3(Hh), 0, stream>>>(
        w_f, w_agg, node_w1 + (size_t)t * 2 * Hh * Hh, node_b1 + t * Hh,
        node_w2 + (size_t)t * Hh * Hh, node_b2 + t * Hh);
  }

  output_kernel<<<dim3(Bb), dim3(Hh), 0, stream>>>(w_f, mask, out_w1, out_b1,
                                                   out_w2, out_b2, out);
}
